// _SAM_7352984011175
// MI455X (gfx1250) — matrix operand order repaired and checked by reading, not compiled, not run on hardware
//
#include <hip/hip_runtime.h>
#include <hip/hip_bf16.h>

#define DEV static __device__ __forceinline__

typedef unsigned short u16t;
typedef __attribute__((ext_vector_type(16))) unsigned short us16;
typedef __attribute__((ext_vector_type(16))) __bf16 v16bf;
typedef __attribute__((ext_vector_type(8)))  float  v8f;

#define NH 12
#define HD 64
#define SQ 4096          // sequence length (64*64)
#define DIM 768
#define GRID 64          // H == W
#define SCALE 0.125f     // HD^-0.5
#define LOG2E 1.44269504088896340736f

// ---------- bf16 helpers ----------
DEV u16t f2bf(float f) {               // round-to-nearest-even (cold paths)
    union { float f; unsigned u; } a; a.f = f;
    return (u16t)((a.u + 0x7FFFu + ((a.u >> 16) & 1u)) >> 16);
}
DEV u16t f2bf_t(float f) {             // truncation (hot path: P probs)
    union { float f; unsigned u; } a; a.f = f;
    return (u16t)(a.u >> 16);
}
DEV v16bf cast16(us16 v) { return __builtin_bit_cast(v16bf, v); }

// A-operand (16x32 bf16): lane holds row m = lane&15; element e:
//   k = 16g+e, g = lane>>4.  One contiguous 16-elem run.
DEV v16bf a_from_bf(const u16t* __restrict__ p, int g) {
    us16 v; const u16t* p0 = p + 16 * g;
#pragma unroll
    for (int e = 0; e < 16; ++e) v[e] = p0[e];
    return cast16(v);
}
// B-operand (32x16 bf16): lane holds col n = lane&15; element e: k = 16g+e.
DEV v16bf b_from_bf(const u16t* __restrict__ p, int g) {
    us16 v; const u16t* q = p + 16 * g;
#pragma unroll
    for (int e = 0; e < 16; ++e) v[e] = q[e];
    return cast16(v);
}
DEV v8f wmma_bf16(v16bf a, v16bf b, v8f c) {
    return __builtin_amdgcn_wmma_f32_16x16x32_bf16(false, a, false, b,
                                                   (short)0, c, false, false);
}

// ============================================================
// Kernel 0: one-time fp32 -> bf16 conversion (RNE), zero-padded tail.
// ============================================================
__global__ __launch_bounds__(256, 1)
void cvt_kernel(const float* __restrict__ src, u16t* __restrict__ dst,
                int nsrc, int ndst) {
    int idx = (blockIdx.x * 256 + threadIdx.x) * 8;
    if (idx >= ndst) return;
    unsigned pk[4];
#pragma unroll
    for (int e = 0; e < 4; ++e) {
        float f0 = (idx + 2 * e     < nsrc) ? src[idx + 2 * e]     : 0.f;
        float f1 = (idx + 2 * e + 1 < nsrc) ? src[idx + 2 * e + 1] : 0.f;
        pk[e] = (unsigned)f2bf(f0) | ((unsigned)f2bf(f1) << 16);
    }
    uint4* d = (uint4*)(dst + idx);
    *d = make_uint4(pk[0], pk[1], pk[2], pk[3]);
}

// ============================================================
// Software-pipelined 64xDIMx64 GEMM body shared by qkv/proj kernels.
// Ping-pong operand buffers in a 2x-unrolled K loop: no register copies.
// ============================================================
DEV void gemm_strip(const u16t* __restrict__ arow,   // A row base (row m0+ln)
                    const u16t* __restrict__ bcol0,  // B col base (col n0+ln)
                    int lh, v8f acc[4]) {
    v16bf a0, b0[4], a1, b1[4];
    a0 = a_from_bf(arow, lh);
#pragma unroll
    for (int nt = 0; nt < 4; ++nt)
        b0[nt] = b_from_bf(bcol0 + (size_t)nt * 16 * DIM, lh);

    for (int kk = 0; kk < DIM; kk += 64) {
        // load odd-slice operands (kk+32 < DIM always for DIM=768)
        a1 = a_from_bf(arow + kk + 32, lh);
#pragma unroll
        for (int nt = 0; nt < 4; ++nt)
            b1[nt] = b_from_bf(bcol0 + (size_t)nt * 16 * DIM + kk + 32, lh);
        // compute even slice
#pragma unroll
        for (int nt = 0; nt < 4; ++nt) acc[nt] = wmma_bf16(a0, b0[nt], acc[nt]);
        // reload even-slice operands for next iteration
        if (kk + 64 < DIM) {
            a0 = a_from_bf(arow + kk + 64, lh);
#pragma unroll
            for (int nt = 0; nt < 4; ++nt)
                b0[nt] = b_from_bf(bcol0 + (size_t)nt * 16 * DIM + kk + 64, lh);
        }
        // compute odd slice
#pragma unroll
        for (int nt = 0; nt < 4; ++nt) acc[nt] = wmma_bf16(a1, b1[nt], acc[nt]);
    }
}

// ============================================================
// Kernel 1: QKV = X @ Wqkv^T + b  ->  bf16 Q[h][s][d], K[h][s][d], V^T[h][d][s]
// ============================================================
__global__ __launch_bounds__(128, 1)
void qkv_kernel(const u16t* __restrict__ xb, const u16t* __restrict__ wb,
                const float* __restrict__ bias,
                u16t* __restrict__ qb, u16t* __restrict__ kb,
                u16t* __restrict__ vTb) {
    const int tid = threadIdx.x, wave = tid >> 5, lane = tid & 31;
    const int lh = lane >> 4, ln = lane & 15;
    const int m0 = blockIdx.x * 64 + wave * 16;
    const int n0 = blockIdx.y * 64;

    v8f acc[4]; v8f z = {};
#pragma unroll
    for (int t = 0; t < 4; ++t) acc[t] = z;

    gemm_strip(xb + (size_t)(m0 + ln) * DIM,
               wb + (size_t)(n0 + ln) * DIM, lh, acc);

    const int t = blockIdx.y / NH;        // 0=q 1=k 2=v
    const int h = blockIdx.y % NH;
#pragma unroll
    for (int nt = 0; nt < 4; ++nt) {
#pragma unroll
        for (int i = 0; i < 8; ++i) {
            int s = m0 + i + 8 * lh;
            int d = nt * 16 + ln;
            float val = acc[nt][i] + bias[n0 + nt * 16 + ln];
            u16t bfv = f2bf(val);
            if (t == 0)      qb[((size_t)h * SQ + s) * HD + d] = bfv;
            else if (t == 1) kb[((size_t)h * SQ + s) * HD + d] = bfv;
            else             vTb[((size_t)h * HD + d) * SQ + s] = bfv;
        }
    }
}

// ============================================================
// Kernel 2: fused flash attention + decomposed rel-pos bias.
// grid (hq=64, head=12), block 128 (4 waves); wave owns 16 queries.
// ============================================================
#define TSTRIDE 196     // floats per T row (192 used, padded)
#define PROW    72      // u16 per P row (64 used, padded, 16B aligned runs)

__global__ __launch_bounds__(128, 1)
void attn_kernel(const u16t* __restrict__ qb, const u16t* __restrict__ kb,
                 const u16t* __restrict__ vTb,
                 const u16t* __restrict__ relhb,   // [127][64] bf16
                 const u16t* __restrict__ relwb,   // [128][64] bf16, row127 = 0
                 u16t* __restrict__ ob) {
    __shared__ __align__(16) float Tlds[64 * TSTRIDE];
    __shared__ __align__(16) u16t  Plds[4 * 16 * PROW];

    const int tid = threadIdx.x, wave = tid >> 5, lane = tid & 31;
    const int lh = lane >> 4, ln = lane & 15;
    const int hq = blockIdx.x, h = blockIdx.y;

    // ---- load this wave's 16x64 Q strip as two A operands (reused all phases)
    const u16t* qbase = qb + ((size_t)h * SQ + hq * 64) * HD;
    v16bf aq[2];
#pragma unroll
    for (int ks = 0; ks < 2; ++ks)
        aq[ks] = a_from_bf(qbase + (size_t)(wave * 16 + ln) * HD + 32 * ks, lh);

    // ---- Phase T: T[wq][c] ; c<64: q.rel_pos_h[hq+c] ; c>=64: q.rel_pos_w[c-64]
    v8f z = {};
#pragma unroll
    for (int ct = 0; ct < 12; ++ct) {
        v8f tc = z;
        int c = ct * 16 + ln;
        const u16t* rp = (ct < 4) ? (relhb + (size_t)(hq + c) * HD)
                                  : (relwb + (size_t)(c - 64) * HD);  // row127=0
#pragma unroll
        for (int ks = 0; ks < 2; ++ks)
            tc = wmma_bf16(aq[ks], b_from_bf(rp + 32 * ks, lh), tc);
#pragma unroll
        for (int i = 0; i < 8; ++i)
            Tlds[(wave * 16 + i + 8 * lh) * TSTRIDE + ct * 16 + ln] = tc[i];
    }
    __syncthreads();

    // ---- flash loop over 64 key tiles
    float rowm[8], rowl[8];
    v8f oacc[4];
#pragma unroll
    for (int i = 0; i < 8; ++i) { rowm[i] = -3.0e38f; rowl[i] = 0.f; }
#pragma unroll
    for (int nt = 0; nt < 4; ++nt) oacc[nt] = z;

    const u16t* khead = kb + (size_t)h * SQ * HD;
    const u16t* vhead = vTb + (size_t)h * HD * SQ;

    for (int kt = 0; kt < 64; ++kt) {
        // prefetch next key/value tile into cache while we compute this one
        if (kt + 1 < 64) {
            int r = tid >> 1, half = (tid & 1) * 32;
            __builtin_prefetch(khead + (size_t)((kt + 1) * 64 + r) * HD + half, 0, 1);
            __builtin_prefetch(vhead + (size_t)r * SQ + (kt + 1) * 64 + half, 0, 1);
        }
        // ---- batch-load all K B-operands, then run score WMMAs
        v16bf bk[4][2];
#pragma unroll
        for (int nt = 0; nt < 4; ++nt) {
            const u16t* kp = khead + (size_t)(kt * 64 + nt * 16 + ln) * HD;
#pragma unroll
            for (int ks = 0; ks < 2; ++ks) bk[nt][ks] = b_from_bf(kp + 32 * ks, lh);
        }
        v8f sacc[4];
#pragma unroll
        for (int nt = 0; nt < 4; ++nt) sacc[nt] = z;
#pragma unroll
        for (int nt = 0; nt < 4; ++nt)
#pragma unroll
            for (int ks = 0; ks < 2; ++ks)
                sacc[nt] = wmma_bf16(aq[ks], bk[nt][ks], sacc[nt]);

        // ---- issue V B-operand loads now; they fly during softmax + LDS trip
        v16bf bv[4][2];
#pragma unroll
        for (int nt = 0; nt < 4; ++nt) {
            const u16t* vp = vhead + (size_t)(nt * 16 + ln) * SQ + kt * 64;
#pragma unroll
            for (int ks = 0; ks < 2; ++ks) bv[nt][ks] = b_from_bf(vp + 32 * ks, lh);
        }

        // ---- scale + rel-pos bias, move to log2 domain
        float s2[4][8];
#pragma unroll
        for (int i = 0; i < 8; ++i) {
            int wq = wave * 16 + i + 8 * lh;
            float th = Tlds[wq * TSTRIDE + (63 - kt)];
#pragma unroll
            for (int nt = 0; nt < 4; ++nt) {
                int kw = nt * 16 + ln;
                float tw = Tlds[wq * TSTRIDE + 127 + wq - kw];
                s2[nt][i] = (sacc[nt][i] * SCALE + th + tw) * LOG2E;
            }
        }
        // ---- online softmax (row stats across the 16-lane n-group)
        float pm[8];
#pragma unroll
        for (int i = 0; i < 8; ++i)
            pm[i] = fmaxf(fmaxf(s2[0][i], s2[1][i]), fmaxf(s2[2][i], s2[3][i]));
#pragma unroll
        for (int mk = 1; mk <= 8; mk <<= 1)
#pragma unroll
            for (int i = 0; i < 8; ++i) pm[i] = fmaxf(pm[i], __shfl_xor(pm[i], mk));
        float alpha[8], ps[8];
#pragma unroll
        for (int i = 0; i < 8; ++i) {
            float mn = fmaxf(rowm[i], pm[i]);
            alpha[i] = exp2f(rowm[i] - mn);
            rowm[i] = mn; ps[i] = 0.f;
        }
#pragma unroll
        for (int nt = 0; nt < 4; ++nt)
#pragma unroll
            for (int i = 0; i < 8; ++i) {
                float p = exp2f(s2[nt][i] - rowm[i]);
                s2[nt][i] = p; ps[i] += p;
            }
#pragma unroll
        for (int mk = 1; mk <= 8; mk <<= 1)
#pragma unroll
            for (int i = 0; i < 8; ++i) ps[i] += __shfl_xor(ps[i], mk);
#pragma unroll
        for (int i = 0; i < 8; ++i) rowl[i] = rowl[i] * alpha[i] + ps[i];
#pragma unroll
        for (int nt = 0; nt < 4; ++nt)
#pragma unroll
            for (int i = 0; i < 8; ++i) oacc[nt][i] *= alpha[i];

        // ---- re-layout P (C/D -> A operand) through wave-private LDS (trunc)
        u16t* pw = Plds + wave * 16 * PROW;
#pragma unroll
        for (int nt = 0; nt < 4; ++nt)
#pragma unroll
            for (int i = 0; i < 8; ++i)
                pw[(i + 8 * lh) * PROW + nt * 16 + ln] = f2bf_t(s2[nt][i]);

        v16bf ap[2];
        const u16t* pr = pw + ln * PROW;
#pragma unroll
        for (int ks = 0; ks < 2; ++ks) ap[ks] = a_from_bf(pr + 32 * ks, lh);

        // ---- O += P @ V
#pragma unroll
        for (int nt = 0; nt < 4; ++nt)
#pragma unroll
            for (int ks = 0; ks < 2; ++ks)
                oacc[nt] = wmma_bf16(ap[ks], bv[nt][ks], oacc[nt]);
    }

    // normalize + store to concatenated layout O[s][h*64+d] (bf16)
#pragma unroll
    for (int nt = 0; nt < 4; ++nt)
#pragma unroll
        for (int i = 0; i < 8; ++i) {
            int wq = wave * 16 + i + 8 * lh;
            int s = hq * 64 + wq;
            int col = h * HD + nt * 16 + ln;
            ob[(size_t)s * DIM + col] = f2bf(oacc[nt][i] / rowl[i]);
        }
}

// ============================================================
// Kernel 3: out = O @ Wproj^T + b   (fp32 out)
// ============================================================
__global__ __launch_bounds__(128, 1)
void proj_kernel(const u16t* __restrict__ ob, const u16t* __restrict__ wb,
                 const float* __restrict__ bias, float* __restrict__ out) {
    const int tid = threadIdx.x, wave = tid >> 5, lane = tid & 31;
    const int lh = lane >> 4, ln = lane & 15;
    const int m0 = blockIdx.x * 64 + wave * 16;
    const int n0 = blockIdx.y * 64;

    v8f acc[4]; v8f z = {};
#pragma unroll
    for (int t = 0; t < 4; ++t) acc[t] = z;

    gemm_strip(ob + (size_t)(m0 + ln) * DIM,
               wb + (size_t)(n0 + ln) * DIM, lh, acc);

#pragma unroll
    for (int nt = 0; nt < 4; ++nt)
#pragma unroll
        for (int i = 0; i < 8; ++i) {
            int s = m0 + i + 8 * lh;
            int col = n0 + nt * 16 + ln;
            out[(size_t)s * DIM + col] = acc[nt][i] + bias[col];
        }
}

// ============================================================
extern "C" void kernel_launch(void* const* d_in, const int* in_sizes, int n_in,
                              void* d_out, int out_size, void* d_ws, size_t ws_size,
                              hipStream_t stream) {
    const float* x      = (const float*)d_in[0];
    const float* qkv_w  = (const float*)d_in[1];
    const float* qkv_b  = (const float*)d_in[2];
    const float* proj_w = (const float*)d_in[3];
    const float* proj_b = (const float*)d_in[4];
    const float* rel_h  = (const float*)d_in[5];
    const float* rel_w  = (const float*)d_in[6];
    float* out = (float*)d_out;

    const size_t SEG = (size_t)NH * SQ * HD * sizeof(u16t);  // 6 MB each
    char* ws = (char*)d_ws;
    u16t* qb     = (u16t*)(ws);
    u16t* kbuf   = (u16t*)(ws + SEG);
    u16t* vTb    = (u16t*)(ws + 2 * SEG);
    u16t* obuf   = (u16t*)(ws + 3 * SEG);
    u16t* xb     = (u16t*)(ws + 4 * SEG);                       // 4096*768
    u16t* wqkvb  = (u16t*)(ws + 5 * SEG);                       // 2304*768
    u16t* wprojb = (u16t*)(ws + 5 * SEG + (size_t)3 * DIM * DIM * 2);
    u16t* relhb  = (u16t*)((char*)wprojb + (size_t)DIM * DIM * 2); // 127*64
    u16t* relwb  = relhb + 127 * HD;                               // 128*64 (pad)

    const int NX = SQ * DIM, NW1 = 3 * DIM * DIM, NW2 = DIM * DIM;
    const int NRH = 127 * HD, NRW_D = 128 * HD;
    cvt_kernel<<<(NX / 8 + 255) / 256, 256, 0, stream>>>(x, xb, NX, NX);
    cvt_kernel<<<(NW1 / 8 + 255) / 256, 256, 0, stream>>>(qkv_w, wqkvb, NW1, NW1);
    cvt_kernel<<<(NW2 / 8 + 255) / 256, 256, 0, stream>>>(proj_w, wprojb, NW2, NW2);
    cvt_kernel<<<(NRH / 8 + 255) / 256, 256, 0, stream>>>(rel_h, relhb, NRH, NRH);
    cvt_kernel<<<(NRW_D / 8 + 255) / 256, 256, 0, stream>>>(rel_w, relwb, NRH, NRW_D);

    qkv_kernel<<<dim3(SQ / 64, (3 * DIM) / 64), 128, 0, stream>>>(
        xb, wqkvb, qkv_b, qb, kbuf, vTb);
    attn_kernel<<<dim3(GRID, NH), 128, 0, stream>>>(
        qb, kbuf, vTb, relhb, relwb, obuf);
    proj_kernel<<<dim3(SQ / 64, DIM / 64), 128, 0, stream>>>(
        obuf, wprojb, proj_b, out);
}
